// Attention_82815559401739
// MI455X (gfx1250) — 16-bit and 8-bit matrix operand order repaired and checked by reading, compiled, not run on hardware
//
#include <hip/hip_runtime.h>

typedef __attribute__((ext_vector_type(16))) _Float16 v16h;
typedef __attribute__((ext_vector_type(8)))  float    v8f;
typedef __attribute__((ext_vector_type(8)))  int      v8i;

#define B_   32
#define N_   197
#define NP_  208
#define C_   768
#define H_   12
#define DH_  64
#define C3_  2304
#define ZP_  128.0f

__global__ __launch_bounds__(256) void k_quant_w(
    const float* __restrict__ wqkv, const float* __restrict__ sqkv,
    const float* __restrict__ wproj, const float* __restrict__ sproj,
    signed char* __restrict__ wq_qkv, signed char* __restrict__ wq_proj) {
  long i = (long)blockIdx.x * 256 + threadIdx.x;
  const long n1 = (long)C3_ * C_;
  const long n2 = (long)C_ * C_;
  if (i < n1) {
    int row = (int)(i / C_);
    float v = rintf(wqkv[i] / sqkv[row]);
    wq_qkv[i] = (signed char)fminf(fmaxf(v, -128.f), 127.f);
  } else if (i < n1 + n2) {
    long j = i - n1;
    int row = (int)(j / C_);
    float v = rintf(wproj[j] / sproj[row]);
    wq_proj[j] = (signed char)fminf(fmaxf(v, -128.f), 127.f);
  }
}

__global__ __launch_bounds__(256) void k_qkv(
    const float* __restrict__ x, const signed char* __restrict__ wq,
    const float* __restrict__ sw, const float* __restrict__ bias,
    const float* __restrict__ s_a1p, signed char* __restrict__ qkv_q) {
  __shared__ __attribute__((aligned(16))) _Float16 Alds[128][32];
  __shared__ __attribute__((aligned(16))) _Float16 Blds[128][32];
  const int mb = blockIdx.x / 18;
  const int cb = blockIdx.x % 18;
  const int t  = threadIdx.x;
  const int w  = t >> 5;
  const int L  = t & 31;
  const int lm = L & 15;
  const int hi = L >> 4;
  const float s_a1 = s_a1p[0];
  const int m0 = mb * 128, c0 = cb * 128;

  v8f acc[8];
#pragma unroll
  for (int nt = 0; nt < 8; nt++)
#pragma unroll
    for (int e = 0; e < 8; e++) acc[nt][e] = 0.f;

  for (int k0 = 0; k0 < C_; k0 += 32) {
    {
      int m  = t >> 1;
      int kh = (t & 1) * 16;
      int row = m0 + m; if (row > B_ * N_ - 1) row = B_ * N_ - 1;
      const float4* s4 = (const float4*)(x + (long)row * C_ + k0 + kh);
      if (k0 + 32 < C_) __builtin_prefetch((const void*)(s4 + 8), 0, 0);
#pragma unroll
      for (int q4 = 0; q4 < 4; q4++) {
        float4 f = s4[q4];
        Alds[m][kh + 4 * q4 + 0] = (_Float16)f.x;
        Alds[m][kh + 4 * q4 + 1] = (_Float16)f.y;
        Alds[m][kh + 4 * q4 + 2] = (_Float16)f.z;
        Alds[m][kh + 4 * q4 + 3] = (_Float16)f.w;
      }
      const int* bsrc = (const int*)(wq + (long)(c0 + m) * C_ + k0 + kh);
      if (k0 + 32 < C_) __builtin_prefetch((const void*)(bsrc + 8), 0, 0);
#pragma unroll
      for (int q4 = 0; q4 < 4; q4++) {
        int pk = bsrc[q4];
        Blds[m][kh + 4 * q4 + 0] = (_Float16)(float)(signed char)(pk & 0xff);
        Blds[m][kh + 4 * q4 + 1] = (_Float16)(float)(signed char)((pk >> 8) & 0xff);
        Blds[m][kh + 4 * q4 + 2] = (_Float16)(float)(signed char)((pk >> 16) & 0xff);
        Blds[m][kh + 4 * q4 + 3] = (_Float16)(float)(signed char)((pk >> 24) & 0xff);
      }
    }
    __syncthreads();
    v16h a;
#pragma unroll
    for (int i = 0; i < 8; i++) {
      int k = ((i < 4) ? 2 * i : 8 + 2 * i) + 8 * hi;
      a[2 * i]     = Alds[w * 16 + lm][k];
      a[2 * i + 1] = Alds[w * 16 + lm][k + 1];
    }
#pragma unroll
    for (int nt = 0; nt < 8; nt++) {
      v16h b;
      int n = nt * 16 + lm;
#pragma unroll
      for (int i = 0; i < 8; i++) {
        int k = ((i < 4) ? 2 * i : 8 + 2 * i) + 8 * hi;
        b[2 * i]     = Blds[n][k];
        b[2 * i + 1] = Blds[n][k + 1];
      }
      acc[nt] = __builtin_amdgcn_wmma_f32_16x16x32_f16(
          false, a, false, b, (short)0, acc[nt], false, false);
    }
    __syncthreads();
  }

#pragma unroll
  for (int nt = 0; nt < 8; nt++) {
    int col = c0 + nt * 16 + lm;
    float scale = sw[col], bb = bias[col];
    int which = col / C_;
    int rem   = col % C_;
    int head  = rem >> 6;
    int d     = rem & 63;
#pragma unroll
    for (int v = 0; v < 8; v++) {
      int row = m0 + w * 16 + v + 8 * hi;
      if (row < B_ * N_) {
        float h = acc[nt][v] * scale + bb;
        float qv = fminf(fmaxf(rintf(h / s_a1) + ZP_, 0.f), 255.f) - ZP_;
        int b = row / N_, n = row % N_;
        long off = ((((long)which * B_ + b) * H_ + head) * NP_ + n) * DH_ + d;
        qkv_q[off] = (signed char)qv;
      }
    }
  }
}

__global__ __launch_bounds__(256) void k_attn(
    const signed char* __restrict__ qkv_q,
    const float* __restrict__ s_a1p, const float* __restrict__ s_attnp,
    const float* __restrict__ s_a2p, signed char* __restrict__ o_q) {
  __shared__ __attribute__((aligned(16))) signed char sc[8][16][224];
  __shared__ __attribute__((aligned(16))) signed char Vt[64][224];
  const int bh = blockIdx.x;
  const int b = bh / H_, h = bh % H_;
  const int t = threadIdx.x, w = t >> 5, L = t & 31, lm = L & 15, hi = L >> 4;
  const long plane = (long)NP_ * DH_;
  const signed char* qg = qkv_q + ((long)(0 * B_ + b) * H_ + h) * plane;
  const signed char* kg = qkv_q + ((long)(1 * B_ + b) * H_ + h) * plane;
  const signed char* vg = qkv_q + ((long)(2 * B_ + b) * H_ + h) * plane;
  const float s_a1 = s_a1p[0], s_attn = s_attnp[0], s_a2 = s_a2p[0];
  const float sc_scale = s_a1 * s_a1 * 0.125f;
  const float x0   = floorf(-0.6931f / s_attn);
  const float bint = floorf(0.96963238f / 0.35815147f / s_attn);
  const float cint = floorf(1.0f / (0.35815147f * s_attn * s_attn));

  for (int idx = t; idx < 64 * 224; idx += 256) {
    int d = idx / 224, key = idx % 224;
    Vt[d][key] = (key < N_) ? vg[(long)key * DH_ + d] : (signed char)0;
  }
  __syncthreads();

  for (int round = 0; round < 2; round++) {
    const int rt = w + 8 * round;
    const bool active = (rt < 13);

    if (active) {
      int qrow = rt * 16 + lm;
      v8i a;
#pragma unroll
      for (int i = 0; i < 8; i++) {
        int k = (i & 1) * 4 + ((i >> 1) & 1) * 16 + (i >> 2) * 32 + 8 * hi;
        a[i] = *(const int*)(qg + (long)qrow * DH_ + k);
      }
      for (int ct = 0; ct < 13; ct++) {
        v8i bf;
        int key = ct * 16 + lm;
#pragma unroll
        for (int i = 0; i < 8; i++) {
          int k = (i & 1) * 4 + ((i >> 1) & 1) * 16 + (i >> 2) * 32 + 8 * hi;
          bf[i] = *(const int*)(kg + (long)key * DH_ + k);
        }
        v8i c;
#pragma unroll
        for (int e = 0; e < 8; e++) c[e] = 0;
        c = __builtin_amdgcn_wmma_i32_16x16x64_iu8(true, a, true, bf, c,
                                                   false, false);
#pragma unroll
        for (int v = 0; v < 8; v++) {
          int m = v + 8 * hi;
          float sf = (float)c[v] * sc_scale;
          float xq = fminf(fmaxf(rintf(sf / s_attn) + ZP_, 0.f), 255.f) - ZP_;
          sc[w][m][ct * 16 + lm] = (signed char)xq;
        }
      }
    }
    __syncthreads();

    if (active) {
      for (int r = 0; r < 16; r++) {
        float mx = -1e30f;
        for (int c2 = L; c2 < N_; c2 += 32)
          mx = fmaxf(mx, (float)sc[w][r][c2]);
#pragma unroll
        for (int off = 16; off > 0; off >>= 1)
          mx = fmaxf(mx, __shfl_xor(mx, off, 32));
        float sum = 0.f;
        for (int c2 = L; c2 < N_; c2 += 32) {
          float xi = fmaxf((float)sc[w][r][c2] - mx, 30.f * x0);
          float qf = floorf(xi / x0);
          float rr = xi - x0 * qf;
          float z  = rr * (rr + bint) + cint;
          sum += fmaxf(floorf(ldexpf(z, 30 - (int)qf)), 0.f);
        }
#pragma unroll
        for (int off = 16; off > 0; off >>= 1)
          sum += __shfl_xor(sum, off, 32);
        for (int c2 = L; c2 < 224; c2 += 32) {
          signed char code = 16;
          if (c2 < N_) {
            float xi = fmaxf((float)sc[w][r][c2] - mx, 30.f * x0);
            float qf = floorf(xi / x0);
            float rr = xi - x0 * qf;
            float z  = rr * (rr + bint) + cint;
            float e  = fmaxf(floorf(ldexpf(z, 30 - (int)qf)), 0.f);
            float ratio = rintf(sum / fmaxf(e, 1.f));
            float big = floorf(log2f(ratio));
            big += (ratio - exp2f(big) >= exp2f(big - 1.f)) ? 1.f : 0.f;
            code = (big >= 16.f || e == 0.f)
                       ? (signed char)16
                       : (signed char)fminf(fmaxf(big, 0.f), 15.f);
          }
          sc[w][r][c2] = code;
        }
      }
    }
    __syncthreads();

    if (active) {
      v8f oacc[4];
#pragma unroll
      for (int nt = 0; nt < 4; nt++)
#pragma unroll
        for (int e = 0; e < 8; e++) oacc[nt][e] = 0.f;
      for (int k0 = 0; k0 < 224; k0 += 32) {
        v16h a;
#pragma unroll
        for (int j = 0; j < 16; j++) {
          int k = k0 + j + 16 * hi;
          int code = sc[w][lm][k];
          float val = (code < 16)
                          ? __uint_as_float((unsigned)(127 - code) << 23)
                          : 0.f;
          a[j] = (_Float16)val;
        }
#pragma unroll
        for (int nt = 0; nt < 4; nt++) {
          v16h bf;
          int d = nt * 16 + lm;
#pragma unroll
          for (int i = 0; i < 4; i++) {
            int kk = k0 + 16 * hi + 4 * i;
            int pk = *(const int*)&Vt[d][kk];
            bf[4 * i + 0] = (_Float16)(float)(signed char)(pk & 0xff);
            bf[4 * i + 1] = (_Float16)(float)(signed char)((pk >> 8) & 0xff);
            bf[4 * i + 2] = (_Float16)(float)(signed char)((pk >> 16) & 0xff);
            bf[4 * i + 3] = (_Float16)(float)(signed char)((pk >> 24) & 0xff);
          }
          oacc[nt] = __builtin_amdgcn_wmma_f32_16x16x32_f16(
              false, a, false, bf, (short)0, oacc[nt], false, false);
        }
      }
#pragma unroll
      for (int nt = 0; nt < 4; nt++) {
        int d = nt * 16 + lm;
#pragma unroll
        for (int v = 0; v < 8; v++) {
          int n = rt * 16 + v + 8 * hi;
          if (n < N_) {
            float o = oacc[nt][v] * s_a1;
            float qv = fminf(fmaxf(rintf(o / s_a2) + ZP_, 0.f), 255.f) - ZP_;
            o_q[((long)b * NP_ + n) * C_ + h * DH_ + d] = (signed char)qv;
          }
        }
      }
    }
    __syncthreads();
  }
}

__global__ __launch_bounds__(256) void k_proj(
    const signed char* __restrict__ o_q, const signed char* __restrict__ wq,
    const float* __restrict__ sw, const float* __restrict__ bias,
    const float* __restrict__ s_a2p, const float* __restrict__ s_a3p,
    float* __restrict__ out) {
  __shared__ __attribute__((aligned(16))) signed char Alds[128][64];
  __shared__ __attribute__((aligned(16))) signed char Blds[128][64];
  const int mb = blockIdx.x / 6;
  const int cb = blockIdx.x % 6;
  const int t = threadIdx.x, w = t >> 5, L = t & 31, lm = L & 15, hi = L >> 4;
  const float s_a2 = s_a2p[0], s_a3 = s_a3p[0];
  const int m0 = mb * 128, c0 = cb * 128;

  v8i acc[8];
#pragma unroll
  for (int nt = 0; nt < 8; nt++)
#pragma unroll
    for (int e = 0; e < 8; e++) acc[nt][e] = 0;

  for (int k0 = 0; k0 < C_; k0 += 64) {
    {
      int m  = t >> 1;
      int kh = (t & 1) * 32;
      const int4* asrc = (const int4*)(o_q + (long)(m0 + m) * C_ + k0 + kh);
      if (k0 + 64 < C_) __builtin_prefetch((const void*)(asrc + 4), 0, 0);
      int4* adst = (int4*)&Alds[m][kh];
      adst[0] = asrc[0]; adst[1] = asrc[1];
      const int4* bsrc = (const int4*)(wq + (long)(c0 + m) * C_ + k0 + kh);
      if (k0 + 64 < C_) __builtin_prefetch((const void*)(bsrc + 4), 0, 0);
      int4* bdst = (int4*)&Blds[m][kh];
      bdst[0] = bsrc[0]; bdst[1] = bsrc[1];
    }
    __syncthreads();
    v8i a;
#pragma unroll
    for (int i = 0; i < 8; i++) {
      int k = (i & 1) * 4 + ((i >> 1) & 1) * 16 + (i >> 2) * 32 + 8 * hi;
      a[i] = *(const int*)&Alds[w * 16 + lm][k];
    }
#pragma unroll
    for (int nt = 0; nt < 8; nt++) {
      v8i bf;
#pragma unroll
      for (int i = 0; i < 8; i++) {
        int k = (i & 1) * 4 + ((i >> 1) & 1) * 16 + (i >> 2) * 32 + 8 * hi;
        bf[i] = *(const int*)&Blds[nt * 16 + lm][k];
      }
      acc[nt] = __builtin_amdgcn_wmma_i32_16x16x64_iu8(true, a, true, bf,
                                                       acc[nt], false, false);
    }
    __syncthreads();
  }

#pragma unroll
  for (int nt = 0; nt < 8; nt++) {
    int col = c0 + nt * 16 + lm;
    float scale = s_a2 * sw[col];
    float bb = bias[col];
#pragma unroll
    for (int v = 0; v < 8; v++) {
      int row = m0 + w * 16 + v + 8 * hi;
      int b = row / NP_, n = row % NP_;
      if (n < N_) {
        float val = (float)acc[nt][v] * scale + bb;
        float o = (fminf(fmaxf(rintf(val / s_a3) + ZP_, 0.f), 255.f) - ZP_) *
                  s_a3;
        out[((long)b * N_ + n) * C_ + col] = o;
      }
    }
  }
}

extern "C" void kernel_launch(void* const* d_in, const int* in_sizes, int n_in,
                              void* d_out, int out_size, void* d_ws,
                              size_t ws_size, hipStream_t stream) {
  const float* x       = (const float*)d_in[0];
  const float* w_qkv   = (const float*)d_in[1];
  const float* b_qkv   = (const float*)d_in[2];
  const float* w_proj  = (const float*)d_in[3];
  const float* b_proj  = (const float*)d_in[4];
  const float* sw_qkv  = (const float*)d_in[5];
  const float* sw_proj = (const float*)d_in[6];
  const float* s_a1    = (const float*)d_in[7];
  const float* s_attn  = (const float*)d_in[8];
  const float* s_a2    = (const float*)d_in[9];
  const float* s_a3    = (const float*)d_in[10];
  (void)in_sizes; (void)n_in; (void)out_size; (void)ws_size;

  char* ws = (char*)d_ws;
  signed char* wq_qkv  = (signed char*)(ws + 0);
  signed char* wq_proj = (signed char*)(ws + 1769472);
  signed char* qkv_q   = (signed char*)(ws + 2359296);
  signed char* o_q     = (signed char*)(ws + 17694720);
  float* out = (float*)d_out;

  const int totalw = C3_ * C_ + C_ * C_;
  hipLaunchKernelGGL(k_quant_w, dim3((totalw + 255) / 256), dim3(256), 0,
                     stream, w_qkv, sw_qkv, w_proj, sw_proj, wq_qkv, wq_proj);
  hipLaunchKernelGGL(k_qkv, dim3(50 * 18), dim3(256), 0, stream, x, wq_qkv,
                     sw_qkv, b_qkv, s_a1, qkv_q);
  hipLaunchKernelGGL(k_attn, dim3(B_ * H_), dim3(256), 0, stream, qkv_q, s_a1,
                     s_attn, s_a2, o_q);
  hipLaunchKernelGGL(k_proj, dim3(52 * 6), dim3(256), 0, stream, o_q, wq_proj,
                     sw_proj, b_proj, s_a2, s_a3, out);
}
